// PairLayer_18073222381672
// MI455X (gfx1250) — matrix operand order repaired and checked by reading, not compiled, not run on hardware
//
#include <hip/hip_runtime.h>
#include <hip/hip_bf16.h>
#include <math.h>

// ---------------- problem constants ----------------
#define NNODES   10000
#define NEDGES   131072
#define EDGEDIM  20
#define K1       768      // 6*MUL
#define H1W      512
#define HRW      128
#define KPOST    256
#define MTILE    32       // edges per block (2 WMMA M-tiles per wave)

#define ISQM   0.08838834764831845f   // 1/sqrt(128)
#define ISQ2M  0.0625f                // 1/sqrt(256)
#define ISQ3   0.5773502691896258f    // 1/sqrt(3)
#define EPSLN  1e-5f

typedef __attribute__((ext_vector_type(16))) _Float16 v16h;
typedef __attribute__((ext_vector_type(8)))  float    v8f;

// ---------------- WMMA fragment helpers ----------------
// A fragment (16x32 f16, row-major tile, row stride = K):
// lane<16 : row = lane,    K = {0..7, 16..23}
// lane>=16: row = lane-16, K = {8..15, 24..31}
// caller passes p = base + row*stride + k0 + ((lane<16)?0:8)
__device__ __forceinline__ v16h frag_a(const _Float16* p) {
  v16h r;
#pragma unroll
  for (int i = 0; i < 8; ++i) { r[i] = p[i]; r[i + 8] = p[16 + i]; }
  return r;
}

// B fragment (32x16 f16) from a TRANSPOSED weight matrix Wt[N][K]:
// lane owns column (lane&15); lane<16 holds K={0..7, 16..23}, lane>=16 K={8..15, 24..31}.
// caller passes p = Wt + col*Kstride + k0 + ((lane<16)?0:8)
__device__ __forceinline__ v16h frag_b(const _Float16* p) {
  v16h r;
#pragma unroll
  for (int i = 0; i < 8; ++i) { r[i] = p[i]; r[i + 8] = p[16 + i]; }
  return r;
}

__device__ __forceinline__ v8f wmma_acc(v16h a, v16h b, v8f c) {
  return __builtin_amdgcn_wmma_f32_16x16x32_f16(false, a, false, b, (short)0, c,
                                                false, false);
}

__device__ __forceinline__ float silu_f(float v) {
  return v / (1.f + __expf(-v));
}

// ---------------- kernel 1: weight conversion (f32 -> f16, transposed) ----------------
__global__ __launch_bounds__(256) void prep_weights_kernel(
    const float* __restrict__ Ws1,  const float* __restrict__ Ws2,
    const float* __restrict__ Wr1,  const float* __restrict__ Wr2,
    const float* __restrict__ Wp0,  const float* __restrict__ Wp1,
    const float* __restrict__ Wpre0,const float* __restrict__ Wpre1,
    _Float16* __restrict__ Ws1t,  _Float16* __restrict__ Ws2t,
    _Float16* __restrict__ Wr1t,  _Float16* __restrict__ Wr2t,
    _Float16* __restrict__ Wp0t,  _Float16* __restrict__ Wp1t,
    _Float16* __restrict__ Wpre0t,_Float16* __restrict__ Wpre1t) {
  const int tid = blockIdx.x * blockDim.x + threadIdx.x;
  const int stp = gridDim.x * blockDim.x;
  for (int t = tid; t < 512 * K1; t += stp) {        // Ws1 (768,512) -> [512][768]
    int n = t / K1, k = t % K1;
    Ws1t[t] = (_Float16)Ws1[k * 512 + n];
  }
  for (int t = tid; t < 512 * 512; t += stp) {       // Ws2 (512,512) -> [512][512]
    int n = t >> 9, k = t & 511;
    Ws2t[t] = (_Float16)Ws2[k * 512 + n];
  }
  for (int t = tid; t < 128 * 32; t += stp) {        // Wr1 (20,128) -> [128][32] zero-pad
    int n = t >> 5, k = t & 31;
    Wr1t[t] = (k < EDGEDIM) ? (_Float16)Wr1[k * 128 + n] : (_Float16)0.f;
  }
  for (int t = tid; t < 512 * 128; t += stp) {       // Wr2 (128,512) -> [512][128]
    int n = t >> 7, k = t & 127;
    Wr2t[t] = (_Float16)Wr2[k * 512 + n];
  }
  for (int t = tid; t < 128 * 256; t += stp) {       // Wpost0/1 (256,128) -> [128][256]
    int n = t >> 8, k = t & 255;
    Wp0t[t] = (_Float16)Wp0[k * 128 + n];
    Wp1t[t] = (_Float16)Wp1[k * 128 + n];
  }
  for (int t = tid; t < 128 * 128; t += stp) {       // Wpre0/1 (128,128) -> [128][128]
    int n = t >> 7, k = t & 127;
    Wpre0t[t] = (_Float16)Wpre0[k * 128 + n];
    Wpre1t[t] = (_Float16)Wpre1[k * 128 + n];
  }
}

// ---------------- kernel 2: per-node prep (x0, sj = s@Wpre0, vj = v@Wpre1) ----------------
__global__ __launch_bounds__(256) void node_kernel(
    const float* __restrict__ x, const float* __restrict__ bpre0,
    const _Float16* __restrict__ Wpre0t, const _Float16* __restrict__ Wpre1t,
    _Float16* __restrict__ x0h, float* __restrict__ sj, float* __restrict__ vj) {
  __shared__ _Float16 As[16 * 128];
  __shared__ _Float16 Av[3][16 * 128];

  const int tid = threadIdx.x;
  const int n0 = blockIdx.x * 16;

  for (int i = tid; i < 16 * 128; i += 256) {
    const int m = i >> 7, ch = i & 127;
    const int n = n0 + m;
    const float sv = x[n * 512 + ch];
    As[i] = (_Float16)sv;
    float vsum = 0.f;
#pragma unroll
    for (int c = 0; c < 3; ++c) {
      const float vv = x[n * 512 + 128 + ch * 3 + c];
      Av[c][i] = (_Float16)vv;
      vsum += vv * vv;
    }
    x0h[n * 256 + ch] = (_Float16)(sv * sv);
    x0h[n * 256 + 128 + ch] = (_Float16)vsum;
  }
  __syncthreads();

  const int wv = tid >> 5, lane = tid & 31;
  const int am = lane & 15;
  const int akb = (lane < 16) ? 0 : 8;
  const int bkb = (lane < 16) ? 0 : 8;
  const int nl = lane & 15;
  const int mr0 = (lane < 16) ? 0 : 8;
  const int ncol = wv * 16 + nl;

  {  // sj = s @ Wpre0 * ISQM + bpre0
    v8f acc = {};
#pragma unroll
    for (int k0 = 0; k0 < 128; k0 += 32)
      acc = wmma_acc(frag_a(As + am * 128 + k0 + akb),
                     frag_b(Wpre0t + ncol * 128 + k0 + bkb), acc);
    const float b = bpre0[ncol];
#pragma unroll
    for (int r = 0; r < 8; ++r)
      sj[(n0 + r + mr0) * 128 + ncol] = acc[r] * ISQM + b;
  }
#pragma unroll
  for (int c = 0; c < 3; ++c) {  // vj_c = v_c @ Wpre1 * ISQM
    v8f acc = {};
#pragma unroll
    for (int k0 = 0; k0 < 128; k0 += 32)
      acc = wmma_acc(frag_a(Av[c] + am * 128 + k0 + akb),
                     frag_b(Wpre1t + ncol * 128 + k0 + bkb), acc);
#pragma unroll
    for (int r = 0; r < 8; ++r)
      vj[(n0 + r + mr0) * 384 + ncol * 3 + c] = acc[r] * ISQM;
  }
}

// ---------------- kernel 3: fused edge kernel ----------------
// 32 edges / block, 8 waves; each wave does 2 WMMA M-tiles per B fragment,
// halving L2 B-operand traffic vs 16-edge tiles.
// LDS: three barrier-separated unioned regions, 262400 B total.
__global__ __launch_bounds__(256) void edge_kernel(
    const float* __restrict__ x, const float* __restrict__ ea,
    const int* __restrict__ ei,
    const float* __restrict__ bs1, const float* __restrict__ bs2,
    const float* __restrict__ br1, const float* __restrict__ br2,
    const float* __restrict__ bpost0,
    const float* __restrict__ lnw0, const float* __restrict__ lnb0,
    const float* __restrict__ lnw1,
    const _Float16* __restrict__ Ws1t, const _Float16* __restrict__ Ws2t,
    const _Float16* __restrict__ Wr1t, const _Float16* __restrict__ Wr2t,
    const _Float16* __restrict__ Wp0t, const _Float16* __restrict__ Wp1t,
    const _Float16* __restrict__ x0h, const float* __restrict__ sj,
    const float* __restrict__ vj, float* __restrict__ out) {
  extern __shared__ char smem[];
  // Region A (64 KB): s0h+eah (phase 0-1)  |  twf (phase 2-3)
  _Float16* s0h = (_Float16*)smem;                  // 32 x 768 f16 = 48 KB
  _Float16* eah = (_Float16*)(smem + 49152);        // 32 x 32  f16 =  2 KB
  float*    twf = (float*)smem;                     // 32 x 512 f32 = 64 KB
  // Region B (64 KB): h1h+hrh (phase 1-2)  |  scath+vcath (phase 3-4)
  char* regB = smem + 65536;
  _Float16* h1h   = (_Float16*)regB;                // 32 x 512 f16 = 32 KB
  _Float16* hrh   = (_Float16*)(regB + 32768);      // 32 x 128 f16 =  8 KB
  _Float16* scath = (_Float16*)regB;                // 32 x 256 f16 = 16 KB
  _Float16* vcath = (_Float16*)(regB + 16384);      // 3 x 32 x 256 f16 = 48 KB
  // Region C (128 KB): si/sjd/vi/vjd (phase 0-3)  |  fsf+fvf (phase 4-5)
  char* regC = smem + 131072;
  float* si_f = (float*)regC;                       // 32 x 128 = 16 KB
  float* sjdf = (float*)(regC + 16384);             // 32 x 128 = 16 KB
  float* vi_f = (float*)(regC + 32768);             // 32 x 384 = 48 KB
  float* vjdf = (float*)(regC + 81920);             // 32 x 384 = 48 KB
  float* fsf  = (float*)regC;                       // 32 x 128 = 16 KB
  float* fvf  = (float*)(regC + 16384);             // 32 x 384 = 48 KB
  int* idxs = (int*)(smem + 262144);                // 64 ints

  const int tid = threadIdx.x;
  const int e0 = blockIdx.x * MTILE;

  if (tid < MTILE) {
    idxs[tid] = ei[e0 + tid];                    // src
    idxs[MTILE + tid] = ei[NEDGES + e0 + tid];   // dst
  }
  if (tid == 0) {  // warm L2 with the shared f16 weights
    __builtin_prefetch(Ws1t, 0, 1);
    __builtin_prefetch(Ws2t, 0, 1);
    __builtin_prefetch(Wr2t, 0, 1);
  }
  __syncthreads();

  // ---- Phase 0: gather + build s0 (f16) ----
  for (int i = tid; i < MTILE * 128; i += 256) {
    const int e = i >> 7, ch = i & 127;
    const int sn = idxs[e], dn = idxs[MTILE + e];
    const float siv = x[sn * 512 + ch];
    const float sdv = x[dn * 512 + ch];
    si_f[e * 128 + ch] = siv;
    sjdf[e * 128 + ch] = sj[dn * 128 + ch];
    s0h[e * 768 + ch] = (_Float16)(siv * sdv);
    float dv = 0.f;
#pragma unroll
    for (int c = 0; c < 3; ++c) {
      const float a = x[sn * 512 + 128 + ch * 3 + c];
      const float b = x[dn * 512 + 128 + ch * 3 + c];
      vi_f[e * 384 + ch * 3 + c] = a;
      vjdf[e * 384 + ch * 3 + c] = vj[dn * 384 + ch * 3 + c];
      dv += a * b;
    }
    s0h[e * 768 + 128 + ch] = (_Float16)dv;
  }
  for (int i = tid; i < MTILE * 256; i += 256) {
    const int e = i >> 8, ch = i & 255;
    s0h[e * 768 + 256 + ch] = x0h[idxs[e] * 256 + ch];
    s0h[e * 768 + 512 + ch] = x0h[idxs[MTILE + e] * 256 + ch];
  }
  for (int i = tid; i < MTILE * 32; i += 256) {
    const int e = i >> 5, ch = i & 31;
    eah[i] = (ch < EDGEDIM) ? (_Float16)ea[(e0 + e) * EDGEDIM + ch]
                            : (_Float16)0.f;
  }
  __syncthreads();

  const int wv = tid >> 5, lane = tid & 31;
  const int am = lane & 15;
  const int akb = (lane < 16) ? 0 : 8;
  const int bkb = (lane < 16) ? 0 : 8;
  const int nl = lane & 15;
  const int mr0 = (lane < 16) ? 0 : 8;

  // ---- Phase 1: h1 = silu(s0 @ Ws1 + bs1); hr = silu(ea @ Wr1 + br1) ----
  for (int nt = wv; nt < 32; nt += 8) {
    v8f a0 = {}, a1 = {};
#pragma unroll 4
    for (int k0 = 0; k0 < K1; k0 += 32) {
      const v16h b = frag_b(Ws1t + (nt * 16 + nl) * K1 + k0 + bkb);
      a0 = wmma_acc(frag_a(s0h + am * K1 + k0 + akb), b, a0);
      a1 = wmma_acc(frag_a(s0h + (16 + am) * K1 + k0 + akb), b, a1);
    }
    const int n = nt * 16 + nl;
    const float b = bs1[n];
#pragma unroll
    for (int r = 0; r < 8; ++r) {
      h1h[(r + mr0) * H1W + n] = (_Float16)silu_f(a0[r] + b);
      h1h[(16 + r + mr0) * H1W + n] = (_Float16)silu_f(a1[r] + b);
    }
  }
  {
    const v16h b = frag_b(Wr1t + (wv * 16 + nl) * 32 + bkb);
    v8f a0 = {}, a1 = {};
    a0 = wmma_acc(frag_a(eah + am * 32 + akb), b, a0);
    a1 = wmma_acc(frag_a(eah + (16 + am) * 32 + akb), b, a1);
    const int n = wv * 16 + nl;
    const float bb = br1[n];
#pragma unroll
    for (int r = 0; r < 8; ++r) {
      hrh[(r + mr0) * HRW + n] = (_Float16)silu_f(a0[r] + bb);
      hrh[(16 + r + mr0) * HRW + n] = (_Float16)silu_f(a1[r] + bb);
    }
  }
  __syncthreads();

  // ---- Phase 2: tw = (h1@Ws2 + bs2) * (hr@Wr2 + br2)  (writes over s0h region) ----
  for (int nt = wv; nt < 32; nt += 8) {
    v8f sA0 = {}, sA1 = {}, sR0 = {}, sR1 = {};
#pragma unroll 4
    for (int k0 = 0; k0 < H1W; k0 += 32) {
      const v16h b = frag_b(Ws2t + (nt * 16 + nl) * H1W + k0 + bkb);
      sA0 = wmma_acc(frag_a(h1h + am * H1W + k0 + akb), b, sA0);
      sA1 = wmma_acc(frag_a(h1h + (16 + am) * H1W + k0 + akb), b, sA1);
    }
#pragma unroll
    for (int k0 = 0; k0 < HRW; k0 += 32) {
      const v16h b = frag_b(Wr2t + (nt * 16 + nl) * HRW + k0 + bkb);
      sR0 = wmma_acc(frag_a(hrh + am * HRW + k0 + akb), b, sR0);
      sR1 = wmma_acc(frag_a(hrh + (16 + am) * HRW + k0 + akb), b, sR1);
    }
    const int n = nt * 16 + nl;
    const float b2 = bs2[n], b3 = br2[n];
#pragma unroll
    for (int r = 0; r < 8; ++r) {
      twf[(r + mr0) * 512 + n] = (sA0[r] + b2) * (sR0[r] + b3);
      twf[(16 + r + mr0) * 512 + n] = (sA1[r] + b2) * (sR1[r] + b3);
    }
  }
  __syncthreads();

  // ---- Phase 3: tensor product -> s_cat (32x256), v_cat (3 x 32x256) f16 ----
  for (int i = tid; i < MTILE * 128; i += 256) {
    const int e = i >> 7, ch = i & 127;
    const float w00 = twf[e * 512 + ch];
    const float w01 = twf[e * 512 + 128 + ch];
    const float w10 = twf[e * 512 + 256 + ch];
    const float w11 = twf[e * 512 + 384 + ch];
    const float siv = si_f[e * 128 + ch];
    const float sjd = sjdf[e * 128 + ch];
    scath[e * 256 + ch] = (_Float16)(w00 * siv * sjd);
    float dv = 0.f;
#pragma unroll
    for (int c = 0; c < 3; ++c)
      dv += vi_f[e * 384 + ch * 3 + c] * vjdf[e * 384 + ch * 3 + c];
    scath[e * 256 + 128 + ch] = (_Float16)(w11 * dv * ISQ3);
#pragma unroll
    for (int c = 0; c < 3; ++c) {
      vcath[c * (MTILE * 256) + e * 256 + ch] =
          (_Float16)(w01 * siv * vjdf[e * 384 + ch * 3 + c]);
      vcath[c * (MTILE * 256) + e * 256 + 128 + ch] =
          (_Float16)(w10 * vi_f[e * 384 + ch * 3 + c] * sjd);
    }
  }
  __syncthreads();

  // ---- Phase 4: fs = s_cat@Wpost0*ISQ2M + bpost0 ; fv_c = v_cat_c@Wpost1*ISQ2M ----
  {
    v8f a0 = {}, a1 = {};
#pragma unroll
    for (int k0 = 0; k0 < KPOST; k0 += 32) {
      const v16h b = frag_b(Wp0t + (wv * 16 + nl) * KPOST + k0 + bkb);
      a0 = wmma_acc(frag_a(scath + am * KPOST + k0 + akb), b, a0);
      a1 = wmma_acc(frag_a(scath + (16 + am) * KPOST + k0 + akb), b, a1);
    }
    const int n = wv * 16 + nl;
    const float b = bpost0[n];
#pragma unroll
    for (int r = 0; r < 8; ++r) {
      fsf[(r + mr0) * 128 + n] = a0[r] * ISQ2M + b;
      fsf[(16 + r + mr0) * 128 + n] = a1[r] * ISQ2M + b;
    }
  }
#pragma unroll
  for (int c = 0; c < 3; ++c) {
    v8f a0 = {}, a1 = {};
#pragma unroll
    for (int k0 = 0; k0 < KPOST; k0 += 32) {
      const v16h b = frag_b(Wp1t + (wv * 16 + nl) * KPOST + k0 + bkb);
      a0 = wmma_acc(frag_a(vcath + c * (MTILE * 256) + am * KPOST + k0 + akb),
                    b, a0);
      a1 = wmma_acc(
          frag_a(vcath + c * (MTILE * 256) + (16 + am) * KPOST + k0 + akb), b,
          a1);
    }
    const int n = wv * 16 + nl;
#pragma unroll
    for (int r = 0; r < 8; ++r) {
      fvf[(r + mr0) * 384 + n * 3 + c] = a0[r] * ISQ2M;
      fvf[(16 + r + mr0) * 384 + n * 3 + c] = a1[r] * ISQ2M;
    }
  }
  __syncthreads();

  // ---- Phase 5: LayerNorm(fs), vector-norm(fv), merged write-out ----
#pragma unroll
  for (int eh = 0; eh < 2; ++eh) {
    const int e = eh * 16 + (tid >> 4);  // edge in tile
    const int t = tid & 15;
    float sm = 0.f;
#pragma unroll
    for (int j = 0; j < 8; ++j) sm += fsf[e * 128 + t + j * 16];
#pragma unroll
    for (int off = 8; off; off >>= 1) sm += __shfl_xor(sm, off, 16);
    const float mean = sm * (1.f / 128.f);

    float sq = 0.f;
#pragma unroll
    for (int j = 0; j < 8; ++j) {
      const float d = fsf[e * 128 + t + j * 16] - mean;
      sq += d * d;
    }
#pragma unroll
    for (int off = 8; off; off >>= 1) sq += __shfl_xor(sq, off, 16);
    const float sinv = 1.f / (sqrtf(sq * (1.f / 128.f)) + EPSLN);

    float vs = 0.f;
#pragma unroll
    for (int j = 0; j < 8; ++j) {
      const int ch = t + j * 16;
#pragma unroll
      for (int c = 0; c < 3; ++c) {
        const float vv = fvf[e * 384 + ch * 3 + c];
        vs += vv * vv;
      }
    }
#pragma unroll
    for (int off = 8; off; off >>= 1) vs += __shfl_xor(vs, off, 16);
    const float vinv = 1.f / (sqrtf(vs * (1.f / 384.f)) + EPSLN);

    float* orow = out + (size_t)(e0 + e) * 512;
#pragma unroll
    for (int j = 0; j < 8; ++j) {
      const int ch = t + j * 16;
      orow[ch] = (fsf[e * 128 + ch] - mean) * sinv * lnw0[ch] + lnb0[ch];
      const float lw = lnw1[ch] * vinv;
#pragma unroll
      for (int c = 0; c < 3; ++c)
        orow[128 + ch * 3 + c] = fvf[e * 384 + ch * 3 + c] * lw;
    }
  }
}

// ---------------- host-side launch ----------------
extern "C" void kernel_launch(void* const* d_in, const int* in_sizes, int n_in,
                              void* d_out, int out_size, void* d_ws,
                              size_t ws_size, hipStream_t stream) {
  const float* x     = (const float*)d_in[0];
  const float* ea    = (const float*)d_in[1];
  const int*   ei    = (const int*)d_in[2];
  const float* Wpre0 = (const float*)d_in[3];
  const float* bpre0 = (const float*)d_in[4];
  const float* Wpre1 = (const float*)d_in[5];
  const float* Ws1   = (const float*)d_in[6];
  const float* bs1   = (const float*)d_in[7];
  const float* Ws2   = (const float*)d_in[8];
  const float* bs2   = (const float*)d_in[9];
  const float* Wr1   = (const float*)d_in[10];
  const float* br1   = (const float*)d_in[11];
  const float* Wr2   = (const float*)d_in[12];
  const float* br2   = (const float*)d_in[13];
  const float* Wp0   = (const float*)d_in[14];
  const float* bp0   = (const float*)d_in[15];
  const float* Wp1   = (const float*)d_in[16];
  const float* lnw0  = (const float*)d_in[17];
  const float* lnb0  = (const float*)d_in[18];
  const float* lnw1  = (const float*)d_in[19];
  float* out = (float*)d_out;

  // workspace carve (256B aligned blocks)
  char* ws = (char*)d_ws;
  size_t off = 0;
  auto carve = [&](size_t bytes) -> void* {
    void* p = ws + off;
    off = (off + bytes + 255) & ~(size_t)255;
    return p;
  };
  _Float16* Ws1t   = (_Float16*)carve(512 * K1 * 2);
  _Float16* Ws2t   = (_Float16*)carve(512 * 512 * 2);
  _Float16* Wr1t   = (_Float16*)carve(128 * 32 * 2);
  _Float16* Wr2t   = (_Float16*)carve(512 * 128 * 2);
  _Float16* Wp0t   = (_Float16*)carve(128 * 256 * 2);
  _Float16* Wp1t   = (_Float16*)carve(128 * 256 * 2);
  _Float16* Wpre0t = (_Float16*)carve(128 * 128 * 2);
  _Float16* Wpre1t = (_Float16*)carve(128 * 128 * 2);
  _Float16* x0h    = (_Float16*)carve((size_t)NNODES * 256 * 2);
  float*    sjw    = (float*)carve((size_t)NNODES * 128 * 4);
  float*    vjw    = (float*)carve((size_t)NNODES * 384 * 4);

  prep_weights_kernel<<<512, 256, 0, stream>>>(
      Ws1, Ws2, Wr1, Wr2, Wp0, Wp1, Wpre0, Wpre1, Ws1t, Ws2t, Wr1t, Wr2t,
      Wp0t, Wp1t, Wpre0t, Wpre1t);

  node_kernel<<<NNODES / 16, 256, 0, stream>>>(x, bpre0, Wpre0t, Wpre1t, x0h,
                                               sjw, vjw);

  const size_t smem_bytes = 262400;  // 3 unioned LDS regions (<= 320 KB/WGP)
  edge_kernel<<<NEDGES / MTILE, 256, smem_bytes, stream>>>(
      x, ea, ei, bs1, bs2, br1, br2, bp0, lnw0, lnb0, lnw1, Ws1t, Ws2t, Wr1t,
      Wr2t, Wp0t, Wp1t, x0h, sjw, vjw, out);
}
